// AttentionLayer_53515292508185
// MI455X (gfx1250) — compile-verified, matrix operand order checked by reading, not run on hardware
//
#include <hip/hip_runtime.h>

typedef __bf16 bf16;
typedef __attribute__((ext_vector_type(16))) __bf16 v16bf;
typedef __attribute__((ext_vector_type(8)))  __bf16 v8bf;
typedef __attribute__((ext_vector_type(8)))  float  v8f;

#define PADV (-2.0f)

// Problem dimensions are fixed by the reference; bake them in so every
// K-loop has a compile-time trip count (full unroll -> no register rotation
// copies, scheduler can software-pipeline the load clauses).
constexpr int CB = 4, CN = 2048, CD = 256, CH = 4, CE = 256;
constexpr int CKD = CE * CH;   // 1024

// ---------------------------------------------------------------------------
// WMMA helpers (CDNA5 wave32, 16x16x32 bf16 -> f32)
// ---------------------------------------------------------------------------
__device__ __forceinline__ v8f wmma_bf16(v16bf a, v16bf b, v8f c) {
  // 8 args: (neg_a, A, neg_b, B, c_mod, C, reuse_a, reuse_b)
  return __builtin_amdgcn_wmma_f32_16x16x32_bf16(false, a, false, b, (short)0, c,
                                                 false, false);
}

// Load one 16x32 bf16 tile in A-operand layout from a row-major matrix.
// ISA 7.12.2: lanes 0-15 hold row M=lane, K=0..7 (VGPR0-3) and K=16..23
// (VGPR4-7); lanes 16-31 hold K=8..15 and K=24..31.  The same layout applied
// to B^T rows yields a valid B operand (K x 16, column n contiguous).
__device__ __forceinline__ v16bf load_a16x32(const bf16* p, int ld) {
  int lane = threadIdx.x & 31;
  int row  = lane & 15;
  int kb   = (lane >> 4) << 3;          // 0 or 8
  const bf16* r = p + (size_t)row * ld + kb;
  v8bf lo = *(const v8bf*)(r);          // K = kb .. kb+7
  v8bf hi = *(const v8bf*)(r + 16);     // K = kb+16 .. kb+23
  return __builtin_shufflevector(lo, hi, 0,1,2,3,4,5,6,7,8,9,10,11,12,13,14,15);
}

__device__ __forceinline__ v8f vzero8() {
  v8f z;
#pragma unroll
  for (int i = 0; i < 8; ++i) z[i] = 0.0f;
  return z;
}

// Reductions across the 16 lanes of each wave half (one C-layout row).
__device__ __forceinline__ float red16_max(float v) {
#pragma unroll
  for (int m = 1; m < 16; m <<= 1) v = fmaxf(v, __shfl_xor(v, m, 32));
  return v;
}
__device__ __forceinline__ float red16_sum(float v) {
#pragma unroll
  for (int m = 1; m < 16; m <<= 1) v += __shfl_xor(v, m, 32);
  return v;
}

// ---------------------------------------------------------------------------
// Kernel 0a: pack Wq/Wk/Wv/Wg [H,D,E] f32 -> Wt [4,H,E,D] bf16 (transposed)
// ---------------------------------------------------------------------------
__global__ void wt_kernel(const float* __restrict__ Wq, const float* __restrict__ Wk,
                          const float* __restrict__ Wv, const float* __restrict__ Wg,
                          bf16* __restrict__ Wt) {
  constexpr int H = CH, D = CD, E = CE;
  int idx = blockIdx.x * blockDim.x + threadIdx.x;
  if (idx >= 4 * H * E * D) return;
  int d   = idx % D;
  int e   = (idx / D) % E;
  int h   = (idx / (D * E)) % H;
  int mat = idx / (D * E * H);
  const float* W = (mat == 0) ? Wq : (mat == 1) ? Wk : (mat == 2) ? Wv : Wg;
  Wt[idx] = (bf16)W[((size_t)h * D + d) * E + e];
}

// Kernel 0b: out_w [K=E*H, E] f32 -> owT [E, K] bf16
__global__ void owt_kernel(const float* __restrict__ ow, bf16* __restrict__ owT) {
  constexpr int K = CKD, E = CE;
  int idx = blockIdx.x * blockDim.x + threadIdx.x;
  if (idx >= E * K) return;
  int k = idx % K;
  int e = idx / K;
  owT[idx] = (bf16)ow[(size_t)k * E + e];
}

// ---------------------------------------------------------------------------
// Kernel 1: input LayerNorm over D, emit bf16 x_norm.  One block per row.
// ---------------------------------------------------------------------------
__global__ void ln_rows_kernel(const float* __restrict__ x, const float* __restrict__ g,
                               const float* __restrict__ b, bf16* __restrict__ xn) {
  constexpr int D = CD;
  int row = blockIdx.x;
  int t   = threadIdx.x;
  const float* xr = x + (size_t)row * D;
  float v = xr[t];
  __shared__ float s1[256], s2[256];
  s1[t] = v; s2[t] = v * v;
  __syncthreads();
  for (int s = 128; s > 0; s >>= 1) {
    if (t < s) { s1[t] += s1[t + s]; s2[t] += s2[t + s]; }
    __syncthreads();
  }
  float mean = s1[0] / (float)D;
  float var  = s2[0] / (float)D - mean * mean;
  float inv  = rsqrtf(var + 1e-6f);
  xn[(size_t)row * D + t] = (bf16)((v - mean) * inv * g[t] + b[t]);
}

// ---------------------------------------------------------------------------
// Kernel 2: fused Q/K/V/Gate projections.  4 waves/block, 16x64 per wave.
// K-loop fully unrolled (8 chunks); loads pipeline across chunks via SSA.
// z = mat*H + h.  V is stored transposed [h,b,E,N]; gate gets sigmoid.
// ---------------------------------------------------------------------------
__global__ void __launch_bounds__(128)
qkvg_kernel(const bf16* __restrict__ xn, const bf16* __restrict__ Wt,
            bf16* __restrict__ Q, bf16* __restrict__ K, bf16* __restrict__ Vt,
            bf16* __restrict__ G) {
  constexpr int B = CB, N = CN, D = CD, H = CH, E = CE;
  int wave = threadIdx.x >> 5;
  int lane = threadIdx.x & 31;
  int row0 = (blockIdx.x * 4 + wave) * 16;   // row in [0, B*N)
  int col0 = blockIdx.y * 64;
  int z    = blockIdx.z;                      // mat*H + h
  int mat  = z / H, h = z % H;
  const bf16* A  = xn + (size_t)row0 * D;
  const bf16* Wm = Wt + (size_t)z * E * D;    // [e][d] (= W^T rows)

  v8f acc[4];
#pragma unroll
  for (int j = 0; j < 4; ++j) acc[j] = vzero8();

#pragma unroll
  for (int k = 0; k < D; k += 32) {
    v16bf a = load_a16x32(A + k, D);
#pragma unroll
    for (int j = 0; j < 4; ++j) {
      v16bf bt = load_a16x32(Wm + (size_t)(col0 + 16 * j) * D + k, D);
      acc[j] = wmma_bf16(a, bt, acc[j]);
    }
  }

  int b    = row0 / N;
  int rloc = lane >> 4;
  int cl   = lane & 15;
  size_t hb = (size_t)h * B + b;
#pragma unroll
  for (int j = 0; j < 4; ++j) {
#pragma unroll
    for (int i = 0; i < 8; ++i) {
      int n = (row0 - b * N) + i + 8 * rloc;
      int e = col0 + 16 * j + cl;
      float v = acc[j][i];
      if (mat == 0)      Q [(hb * N + n) * E + e] = (bf16)v;
      else if (mat == 1) K [(hb * N + n) * E + e] = (bf16)v;
      else if (mat == 2) Vt[(hb * E + e) * N + n] = (bf16)v;
      else               G [(hb * N + n) * E + e] = (bf16)(1.0f / (1.0f + __expf(-v)));
    }
  }
}

// ---------------------------------------------------------------------------
// Kernel 3: flash attention + gate + residual + per-head LN + head concat.
// One wave per 16-row Q tile; 8 waves/block; online softmax over N in
// 32-column chunks.  Q tile (8 A-operands, 64 VGPRs) is loop-invariant and
// kept resident in registers; K/V loads pipeline inside the unrolled loops;
// next chunk's K/V rows are prefetched (global_prefetch_b8).
// ---------------------------------------------------------------------------
__global__ void __launch_bounds__(256, 1)
flash_kernel(const bf16* __restrict__ Q, const bf16* __restrict__ Kb,
             const bf16* __restrict__ Vt, const bf16* __restrict__ G,
             const float* __restrict__ x, const float* __restrict__ mask,
             const float* __restrict__ g_res, const float* __restrict__ b_res,
             bf16* __restrict__ cat) {
  constexpr int B = CB, N = CN, E = CE, H = CH;
  __shared__ __align__(16) bf16 Ps[8][16 * 32];

  int wave = threadIdx.x >> 5;
  int lane = threadIdx.x & 31;
  int n16  = N / 16;
  int qt   = blockIdx.x * 8 + wave;           // global q-tile id
  int hb   = qt / n16;                        // h*B + b
  int qrow0 = (qt % n16) * 16;
  int h = hb / B, b = hb % B;

  const bf16*  Qp = Q  + ((size_t)hb * N + qrow0) * E;
  const bf16*  Kp = Kb + (size_t)hb * N * E;
  const bf16*  Vp = Vt + (size_t)hb * E * N;
  const float* mk = mask + (size_t)b * N;

  int rloc = lane >> 4;
  int cl   = lane & 15;
  const float scale = 0.0625f;                // 1/sqrt(256)

  // Loop-invariant Q tile: 8 A-operands resident in registers.
  v16bf qa[8];
#pragma unroll
  for (int ec = 0; ec < 8; ++ec) qa[ec] = load_a16x32(Qp + ec * 32, E);

  float m_run[8], l_run[8];
  v8f o[16];
#pragma unroll
  for (int i = 0; i < 8; ++i) { m_run[i] = -3.0e38f; l_run[i] = 0.0f; }
#pragma unroll
  for (int et = 0; et < 16; ++et) o[et] = vzero8();

  for (int m0 = 0; m0 < N; m0 += 32) {
    // Prefetch next chunk's K rows and a spread of V^T rows into cache
    // while this chunk computes (lowers to global_prefetch_b8).
    if (m0 + 32 < N) {
      __builtin_prefetch((const void*)(Kp + (size_t)(m0 + 32 + lane) * E), 0, 1);
      __builtin_prefetch((const void*)(Vp + (size_t)(lane * 8) * N + m0 + 32), 0, 1);
    }

    // --- S = Q K^T for a 16x32 score tile (two WMMA accumulators) ---
    v8f s0 = vzero8(), s1 = vzero8();
#pragma unroll
    for (int ec = 0; ec < 8; ++ec) {
      v16bf k0 = load_a16x32(Kp + (size_t)m0 * E + ec * 32, E);
      v16bf k1 = load_a16x32(Kp + (size_t)(m0 + 16) * E + ec * 32, E);
      s0 = wmma_bf16(qa[ec], k0, s0);
      s1 = wmma_bf16(qa[ec], k1, s1);
    }

    // key-side mask as additive -1e9
    float kb0 = (mk[m0 + cl]      == PADV) ? -1e9f : 0.0f;
    float kb1 = (mk[m0 + 16 + cl] == PADV) ? -1e9f : 0.0f;

    float p0[8], p1[8];
#pragma unroll
    for (int i = 0; i < 8; ++i) {
      p0[i] = s0[i] * scale + kb0;
      p1[i] = s1[i] * scale + kb1;
    }
    // --- online softmax per row (row i+8*rloc, cols across 16 lanes) ---
#pragma unroll
    for (int i = 0; i < 8; ++i) {
      float tm   = red16_max(fmaxf(p0[i], p1[i]));
      float nm   = fmaxf(m_run[i], tm);
      float corr = __expf(m_run[i] - nm);
      m_run[i]   = nm;
      p0[i] = __expf(p0[i] - nm);
      p1[i] = __expf(p1[i] - nm);
      l_run[i] = l_run[i] * corr + red16_sum(p0[i] + p1[i]);
#pragma unroll
      for (int et = 0; et < 16; ++et) o[et][i] *= corr;
      int r = i + 8 * rloc;
      Ps[wave][r * 32 + cl]      = (bf16)p0[i];
      Ps[wave][r * 32 + 16 + cl] = (bf16)p1[i];
    }
    __syncthreads();
    // --- O += P V : P is a 16x32 A tile; V^T rows contiguous in Vt ---
    v16bf pa = load_a16x32(&Ps[wave][0], 32);
#pragma unroll
    for (int et = 0; et < 16; ++et) {
      v16bf vb = load_a16x32(Vp + (size_t)(et * 16) * N + m0, N);
      o[et] = wmma_bf16(pa, vb, o[et]);
    }
    __syncthreads();
  }

  // --- epilogue: /l, query mask, gate, residual, per-head LN, concat ---
  float qm[8];
#pragma unroll
  for (int i = 0; i < 8; ++i)
    qm[i] = (mk[qrow0 + i + 8 * rloc] == PADV) ? 0.0f : 1.0f;

#pragma unroll
  for (int et = 0; et < 16; ++et) {
#pragma unroll
    for (int i = 0; i < 8; ++i) {
      int n = qrow0 + i + 8 * rloc;
      int e = et * 16 + cl;
      float gate = (float)G[((size_t)hb * N + n) * E + e];
      float vo = (o[et][i] / l_run[i]) * qm[i] * gate
               + x[((size_t)b * N + n) * E + e];       // residual (D == E)
      o[et][i] = vo;
    }
  }
#pragma unroll
  for (int i = 0; i < 8; ++i) {
    float s = 0.0f, q = 0.0f;
#pragma unroll
    for (int et = 0; et < 16; ++et) { float v = o[et][i]; s += v; q += v * v; }
    s = red16_sum(s); q = red16_sum(q);
    float mean = s / (float)E;
    float inv  = rsqrtf(q / (float)E - mean * mean + 1e-6f);
    int n = qrow0 + i + 8 * rloc;
#pragma unroll
    for (int et = 0; et < 16; ++et) {
      int e = et * 16 + cl;
      float y = (o[et][i] - mean) * inv * g_res[e] + b_res[e];
      cat[((size_t)(b * N + n)) * (size_t)(E * H) + (size_t)e * H + h] = (bf16)y;
    }
  }
}

// ---------------------------------------------------------------------------
// Kernel 4: output projection [BN,1024] x [1024,256] + bias + residual.
// K-loop unrolled 4x (compile-time trip count) for load/WMMA overlap.
// ---------------------------------------------------------------------------
__global__ void __launch_bounds__(128)
outproj_kernel(const bf16* __restrict__ cat, const bf16* __restrict__ owT,
               const float* __restrict__ ob, const float* __restrict__ x,
               float* __restrict__ y) {
  constexpr int Kdim = CKD, E = CE;
  int wave = threadIdx.x >> 5;
  int lane = threadIdx.x & 31;
  int row0 = (blockIdx.x * 4 + wave) * 16;
  int col0 = blockIdx.y * 64;
  const bf16* A = cat + (size_t)row0 * Kdim;

  v8f acc[4];
#pragma unroll
  for (int j = 0; j < 4; ++j) acc[j] = vzero8();

#pragma unroll 4
  for (int k = 0; k < Kdim; k += 32) {
    v16bf a = load_a16x32(A + k, Kdim);
#pragma unroll
    for (int j = 0; j < 4; ++j) {
      v16bf bt = load_a16x32(owT + (size_t)(col0 + 16 * j) * Kdim + k, Kdim);
      acc[j] = wmma_bf16(a, bt, acc[j]);
    }
  }

  int rloc = lane >> 4, cl = lane & 15;
#pragma unroll
  for (int j = 0; j < 4; ++j) {
#pragma unroll
    for (int i = 0; i < 8; ++i) {
      int r = row0 + i + 8 * rloc;
      int c = col0 + 16 * j + cl;
      y[(size_t)r * E + c] = acc[j][i] + ob[c] + x[(size_t)r * E + c];
    }
  }
}

// ---------------------------------------------------------------------------
// Kernel 5: final LayerNorm + output mask.
// ---------------------------------------------------------------------------
__global__ void ln_out_kernel(const float* __restrict__ y, const float* __restrict__ g,
                              const float* __restrict__ bb, const float* __restrict__ mask,
                              float* __restrict__ out) {
  constexpr int D = CE;
  int row = blockIdx.x;
  int t   = threadIdx.x;
  const float* yr = y + (size_t)row * D;
  float v = yr[t];
  __shared__ float s1[256], s2[256];
  s1[t] = v; s2[t] = v * v;
  __syncthreads();
  for (int s = 128; s > 0; s >>= 1) {
    if (t < s) { s1[t] += s1[t + s]; s2[t] += s2[t + s]; }
    __syncthreads();
  }
  float mean = s1[0] / (float)D;
  float var  = s2[0] / (float)D - mean * mean;
  float inv  = rsqrtf(var + 1e-6f);
  float mv   = (mask[row] == PADV) ? 0.0f : 1.0f;
  out[(size_t)row * D + t] = ((v - mean) * inv * g[t] + bb[t]) * mv;
}

// ---------------------------------------------------------------------------
extern "C" void kernel_launch(void* const* d_in, const int* in_sizes, int n_in,
                              void* d_out, int out_size, void* d_ws, size_t ws_size,
                              hipStream_t stream) {
  (void)in_sizes; (void)n_in; (void)out_size; (void)ws_size;
  const float* x     = (const float*)d_in[0];
  const float* mask  = (const float*)d_in[1];
  const float* Wq    = (const float*)d_in[2];
  const float* Wk    = (const float*)d_in[3];
  const float* Wv    = (const float*)d_in[4];
  const float* Wg    = (const float*)d_in[5];
  const float* ow    = (const float*)d_in[6];
  const float* ob    = (const float*)d_in[7];
  const float* g_in  = (const float*)d_in[8];
  const float* b_in  = (const float*)d_in[9];
  const float* g_res = (const float*)d_in[10];
  const float* b_res = (const float*)d_in[11];
  const float* g_out = (const float*)d_in[12];
  const float* b_out = (const float*)d_in[13];

  const int B = CB, N = CN, D = CD, H = CH, E = CE;
  const int BN = B * N;
  const int Kd = CKD;

  // workspace carve-out (bf16 intermediates keep it ~100 MB total)
  char*  ws  = (char*)d_ws;
  size_t off = 0;
  auto carve = [&](size_t bytes) -> void* {
    void* p = ws + off;
    off += (bytes + 255) & ~(size_t)255;
    return p;
  };
  bf16*  xn  = (bf16*)carve((size_t)BN * D * 2);          //  4 MB
  bf16*  Wt  = (bf16*)carve((size_t)4 * H * E * D * 2);   //  2 MB
  bf16*  owT = (bf16*)carve((size_t)E * Kd * 2);          //  0.5 MB
  bf16*  Qb  = (bf16*)carve((size_t)H * B * N * E * 2);   // 16 MB
  bf16*  Kb  = (bf16*)carve((size_t)H * B * N * E * 2);   // 16 MB
  bf16*  Vt  = (bf16*)carve((size_t)H * B * E * N * 2);   // 16 MB
  bf16*  Gb  = (bf16*)carve((size_t)H * B * N * E * 2);   // 16 MB
  bf16*  cat = (bf16*)carve((size_t)BN * Kd * 2);         // 16 MB
  float* yw  = (float*)carve((size_t)BN * E * 4);         //  8 MB

  wt_kernel <<<(4 * H * E * D + 255) / 256, 256, 0, stream>>>(Wq, Wk, Wv, Wg, Wt);
  owt_kernel<<<(E * Kd + 255) / 256,        256, 0, stream>>>(ow, owT);
  ln_rows_kernel<<<BN, 256, 0, stream>>>(x, g_in, b_in, xn);

  dim3 gq(BN / 64, E / 64, 4 * H);
  qkvg_kernel<<<gq, 128, 0, stream>>>(xn, Wt, Qb, Kb, Vt, Gb);

  flash_kernel<<<(H * B * (N / 16)) / 8, 256, 0, stream>>>(Qb, Kb, Vt, Gb, x, mask,
                                                           g_res, b_res, cat);

  dim3 go(BN / 64, E / 64);
  outproj_kernel<<<go, 128, 0, stream>>>(cat, owT, ob, x, yw);

  ln_out_kernel<<<BN, 256, 0, stream>>>(yw, g_out, b_out, mask, (float*)d_out);
}
